// MultiHeadAttention_33852932227543
// MI455X (gfx1250) — matrix operand order repaired and checked by reading, not compiled, not run on hardware
//
#include <hip/hip_runtime.h>

// ---------------------------------------------------------------------------
// MHA forward for gfx1250 (MI455X): bf16 WMMA (v_wmma_f32_16x16x32_bf16),
// f32 accumulation, online-softmax attention, wave32 fragment layouts,
// async-to-LDS double-buffered staging of block-shared operands.
// ---------------------------------------------------------------------------

typedef __attribute__((ext_vector_type(16))) __bf16 v16bf;
typedef __attribute__((ext_vector_type(8)))  __bf16 v8bf;
typedef __attribute__((ext_vector_type(4)))  __bf16 v4bf;
typedef __attribute__((ext_vector_type(8)))  float  v8f;

union BF16x16 { v16bf v; v8bf h[2]; };

#define D_MODEL 1024
#define NHEADS  16
#define HD      64
#define BATCH   2
#define SQ      2048
#define SCACHE  2048
#define STOT    4096      // SCACHE + SQ
#define MROWS   4096      // BATCH * SQ

// ---- gfx1250 async-to-LDS helpers -----------------------------------------
// Per-lane 16B copy Memory -> LDS, tracked by ASYNCcnt (ISA 10.x / 15.18.3).
// LDS destination = low 32 bits of generic shared pointer (aperture layout).
__device__ __forceinline__ unsigned lds_off32(const void* p) {
  return (unsigned)(uintptr_t)p;
}
__device__ __forceinline__ void async_ld16(unsigned lds_off, const void* src) {
  asm volatile("global_load_async_to_lds_b128 %0, %1, off"
               :: "v"(lds_off), "v"(src) : "memory");
}
__device__ __forceinline__ void wait_async0() {
  asm volatile("s_wait_asynccnt 0x0" ::: "memory");
}

__device__ __forceinline__ v8f wmma_bf16(const BF16x16& a, const BF16x16& b, v8f c) {
  return __builtin_amdgcn_wmma_f32_16x16x32_bf16(false, a.v, false, b.v,
                                                 (short)0, c, false, false);
}

// ---------------------------------------------------------------------------
// f32 -> bf16 elementwise convert (vectorized x4)
// ---------------------------------------------------------------------------
__global__ __launch_bounds__(256)
void cvt_f32_bf16_kernel(const float* __restrict__ src, __bf16* __restrict__ dst, int n4) {
  int i = blockIdx.x * blockDim.x + threadIdx.x;
  if (i < n4) {
    float4 v = ((const float4*)src)[i];
    v4bf o = { (__bf16)v.x, (__bf16)v.y, (__bf16)v.z, (__bf16)v.w };
    ((v4bf*)dst)[i] = o;
  }
}

// ---------------------------------------------------------------------------
// W [k][n] f32  ->  Wt [n][k] bf16  (32x32 LDS tile transpose)
// ---------------------------------------------------------------------------
__global__ __launch_bounds__(256)
void transpose_cvt_kernel(const float* __restrict__ W, __bf16* __restrict__ Wt) {
  __shared__ float tile[32][33];
  int x = blockIdx.x * 32 + threadIdx.x;   // source col (n)
  int y = blockIdx.y * 32 + threadIdx.y;   // source row (k)
  #pragma unroll
  for (int i = 0; i < 32; i += 8)
    tile[threadIdx.y + i][threadIdx.x] = W[(size_t)(y + i) * D_MODEL + x];
  __syncthreads();
  int tx = blockIdx.y * 32 + threadIdx.x;  // dest col = k
  int ty = blockIdx.x * 32 + threadIdx.y;  // dest row = n
  #pragma unroll
  for (int i = 0; i < 32; i += 8)
    Wt[(size_t)(ty + i) * D_MODEL + tx] = (__bf16)tile[threadIdx.x][threadIdx.y + i];
}

// ---------------------------------------------------------------------------
// KV-cache: copy f32 cache into d_out K/V (head-major, seq offset 0) and
// produce bf16 K [bh][key][d] + bf16 Vt [bh][d][key] for the attention GEMMs.
// ---------------------------------------------------------------------------
__global__ __launch_bounds__(256)
void cache_kernel(const float* __restrict__ kvk, const float* __restrict__ kvv,
                  float* __restrict__ outK, float* __restrict__ outV,
                  __bf16* __restrict__ Kb, __bf16* __restrict__ Vtb) {
  size_t idx = (size_t)blockIdx.x * blockDim.x + threadIdx.x;
  if (idx >= (size_t)BATCH * NHEADS * SCACHE * HD) return;
  int d = (int)(idx & (HD - 1));
  size_t t = idx >> 6;
  int s  = (int)(t & (SCACHE - 1));
  int bh = (int)(t >> 11);
  float kv = kvk[idx];
  float vv = kvv[idx];
  size_t hm = ((size_t)bh * STOT + s) * HD + d;
  outK[hm] = kv;
  outV[hm] = vv;
  Kb[hm] = (__bf16)kv;
  Vtb[((size_t)bh * HD + d) * STOT + s] = (__bf16)vv;
}

// ---------------------------------------------------------------------------
// GEMM: C[m][n] = A[m][k] * Bt[n][k],  M=4096, N=1024, K=1024, bf16 in, f32 acc.
// Block = 8 waves, each wave a 16(M) x 64(N) strip; the 64-wide Bt strip is
// shared by all 8 waves -> staged once per block into LDS with async loads,
// double-buffered in 128-wide K slabs (16 WMMAs per wave between barriers).
// All 4 B-fragments of a k-step are materialized in distinct registers before
// the WMMA batch so one wait covers them and the WMMAs issue back-to-back.
// ---------------------------------------------------------------------------
#define KSLAB  128
#define BPITCH 136   // elems; 272B row pitch (pads banks for ds_load_b128)

__global__ __launch_bounds__(256)
void gemm_bf16_kernel(const __bf16* __restrict__ A, const __bf16* __restrict__ Bt,
                      __bf16* __restrict__ bf_dst, int bf_mode,
                      float* __restrict__ f_dst, int f_mode) {
  __shared__ __align__(16) __bf16 bs[2][64 * BPITCH];

  const int tid  = threadIdx.x;
  const int lane = tid & 31;
  const int wid  = tid >> 5;
  const int half = lane >> 4;
  const int l16  = lane & 15;
  const int mbase = (blockIdx.x * 8 + wid) * 16;
  const int nbase = blockIdx.y * 64;
  const int koff  = half * 8;

  const __bf16* arow = A + (size_t)(mbase + l16) * D_MODEL;

  // cooperative async stage of one 64 x 128 slab of Bt (16 KB, 4x16B/thread)
  auto stage = [&](int buf, int kb) {
    unsigned base = lds_off32(&bs[buf][0]);
    #pragma unroll
    for (int i = 0; i < 4; ++i) {
      int part = tid + i * 256;
      int row = part >> 4, sub = part & 15;
      async_ld16(base + (unsigned)(row * (BPITCH * 2) + sub * 16),
                 Bt + (size_t)(nbase + row) * D_MODEL + kb + sub * 8);
    }
  };

  stage(0, 0);
  v8f acc[4] = {};
  for (int kb = 0; kb < D_MODEL; kb += KSLAB) {
    const int buf = (kb / KSLAB) & 1;
    wait_async0();
    __syncthreads();
    if (kb + KSLAB < D_MODEL) stage(buf ^ 1, kb + KSLAB);

    const __bf16* bsl = &bs[buf][0];
    #pragma unroll
    for (int ks = 0; ks < KSLAB; ks += 32) {
      __builtin_prefetch(arow + kb + ks + 256, 0, 1);
      BF16x16 a, b0, b1, b2, b3;
      a.h[0] = *(const v8bf*)(arow + kb + ks + half * 16);
      a.h[1] = *(const v8bf*)(arow + kb + ks + half * 16 + 8);
      const __bf16* bl = bsl + (size_t)l16 * BPITCH + ks + half * 16;
      b0.h[0] = *(const v8bf*)(bl);
      b0.h[1] = *(const v8bf*)(bl + 8);
      b1.h[0] = *(const v8bf*)(bl + 16 * BPITCH);
      b1.h[1] = *(const v8bf*)(bl + 16 * BPITCH + 8);
      b2.h[0] = *(const v8bf*)(bl + 32 * BPITCH);
      b2.h[1] = *(const v8bf*)(bl + 32 * BPITCH + 8);
      b3.h[0] = *(const v8bf*)(bl + 48 * BPITCH);
      b3.h[1] = *(const v8bf*)(bl + 48 * BPITCH + 8);
      acc[0] = wmma_bf16(a, b0, acc[0]);
      acc[1] = wmma_bf16(a, b1, acc[1]);
      acc[2] = wmma_bf16(a, b2, acc[2]);
      acc[3] = wmma_bf16(a, b3, acc[3]);
    }
    // next iteration's barrier protects buf from being restaged early
  }

  #pragma unroll
  for (int j = 0; j < 4; ++j) {
    int n = nbase + j * 16 + l16;
    int h = n >> 6, d = n & 63;
    #pragma unroll
    for (int r = 0; r < 8; ++r) {
      int m = mbase + r + half * 8;
      int b = m >> 11, s = m & 2047;
      float v = acc[j][r];
      if (bf_mode == 1)
        bf_dst[(((size_t)(b * 16 + h) * STOT) + SCACHE + s) * HD + d] = (__bf16)v;
      else if (bf_mode == 2)
        bf_dst[(((size_t)(b * 16 + h) * HD) + d) * STOT + SCACHE + s] = (__bf16)v;
      else if (bf_mode == 3)
        bf_dst[(((size_t)(b * 16 + h) * SQ) + s) * HD + d] = (__bf16)v;
      if (f_mode == 1)
        f_dst[(((size_t)(b * 16 + h) * STOT) + SCACHE + s) * HD + d] = v;
      else if (f_mode == 2)
        f_dst[(size_t)m * D_MODEL + n] = v;
    }
  }
}

// ---------------------------------------------------------------------------
// Flash attention: block = 8 waves = 8 consecutive 16-row Q tiles of ONE
// (b,h); the 32-key K/V chunk is identical for all 8 waves -> async-staged
// into LDS once per block, double-buffered. Online softmax in f32.
// ---------------------------------------------------------------------------
#define KPITCH 72    // elems; 144B row pitch for K tile  (32 keys x 64 d)
#define VPITCH 40    // elems;  80B row pitch for Vt tile (64 d x 32 keys)

__global__ __launch_bounds__(256)
void attn_kernel(const __bf16* __restrict__ Qb, const __bf16* __restrict__ Kb,
                 const __bf16* __restrict__ Vtb, __bf16* __restrict__ attn_out) {
  __shared__ __align__(16) __bf16 kt[2][32 * KPITCH];
  __shared__ __align__(16) __bf16 vt[2][64 * VPITCH];
  __shared__ __align__(16) __bf16 lds_p[8][16 * 40];   // per-wave P tile

  const int tid  = threadIdx.x;
  const int lane = tid & 31;
  const int wid  = tid >> 5;
  const int half = lane >> 4;
  const int l16  = lane & 15;
  const int koff = half * 16;

  const int qt = blockIdx.x * 8 + wid;     // 0..4095 q-tiles
  const int bh = qt >> 7;                  // same for all 8 waves in block
  const int qb = (qt & 127) << 4;

  const __bf16* kbase = Kb  + (size_t)bh * STOT * HD;
  const __bf16* vbase = Vtb + (size_t)bh * HD * STOT;

  // cooperative async stage of one 32-key chunk of K and Vt (4KB + 4KB)
  auto stage = [&](int buf, int ck) {
    unsigned kl = lds_off32(&kt[buf][0]);
    unsigned vl = lds_off32(&vt[buf][0]);
    {
      int row = tid >> 3, sub = tid & 7;   // K: 32 rows x 128B
      async_ld16(kl + (unsigned)(row * (KPITCH * 2) + sub * 16),
                 kbase + (size_t)(ck + row) * HD + sub * 8);
    }
    {
      int row = tid >> 2, sub = tid & 3;   // Vt: 64 rows x 64B
      async_ld16(vl + (unsigned)(row * (VPITCH * 2) + sub * 16),
                 vbase + (size_t)row * STOT + ck + sub * 8);
    }
  };

  // Q A-fragments for the two 32-wide k-steps over HD=64
  const __bf16* qrow = Qb + ((size_t)bh * SQ + qb + l16) * HD;
  BF16x16 aq0, aq1;
  aq0.h[0] = *(const v8bf*)(qrow + koff);
  aq0.h[1] = *(const v8bf*)(qrow + 8 + koff);
  aq1.h[0] = *(const v8bf*)(qrow + 32 + koff);
  aq1.h[1] = *(const v8bf*)(qrow + 40 + koff);

  v8f o[4] = {};
  float mx[8], lsum[8];
  #pragma unroll
  for (int r = 0; r < 8; ++r) { mx[r] = -1e30f; lsum[r] = 0.0f; }

  __bf16* pt = &lds_p[wid][0];

  stage(0, 0);
  for (int ck = 0; ck < STOT; ck += 32) {
    const int buf = (ck >> 5) & 1;
    wait_async0();
    __syncthreads();
    if (ck + 32 < STOT) stage(buf ^ 1, ck + 32);

    // ---- S = Q * K^T for keys [ck, ck+32) ; B-frags from LDS ----
    const __bf16* k0 = &kt[buf][0] + l16 * KPITCH + half * 16;
    const __bf16* k1 = &kt[buf][0] + (16 + l16) * KPITCH + half * 16;
    BF16x16 b00, b01, b10, b11;
    b00.h[0] = *(const v8bf*)(k0);       b00.h[1] = *(const v8bf*)(k0 + 8);
    b01.h[0] = *(const v8bf*)(k0 + 32);  b01.h[1] = *(const v8bf*)(k0 + 40);
    b10.h[0] = *(const v8bf*)(k1);       b10.h[1] = *(const v8bf*)(k1 + 8);
    b11.h[0] = *(const v8bf*)(k1 + 32);  b11.h[1] = *(const v8bf*)(k1 + 40);

    v8f s0 = {}, s1 = {};
    s0 = wmma_bf16(aq0, b00, s0);
    s1 = wmma_bf16(aq0, b10, s1);
    s0 = wmma_bf16(aq1, b01, s0);
    s1 = wmma_bf16(aq1, b11, s1);

    // ---- online softmax (rows live across a 16-lane half, per C-layout) ----
    float nmax[8], rsum[8];
    #pragma unroll
    for (int r = 0; r < 8; ++r) {
      s0[r] *= 0.125f;                     // 1/sqrt(64)
      s1[r] *= 0.125f;
      nmax[r] = fmaxf(s0[r], s1[r]);
    }
    #pragma unroll
    for (int mstep = 1; mstep < 16; mstep <<= 1) {
      #pragma unroll
      for (int r = 0; r < 8; ++r)
        nmax[r] = fmaxf(nmax[r], __shfl_xor(nmax[r], mstep, 32));
    }
    #pragma unroll
    for (int r = 0; r < 8; ++r) {
      float newm  = fmaxf(mx[r], nmax[r]);
      float alpha = __expf(mx[r] - newm);
      mx[r] = newm;
      lsum[r] *= alpha;
      o[0][r] *= alpha; o[1][r] *= alpha; o[2][r] *= alpha; o[3][r] *= alpha;
      float p0 = __expf(s0[r] - newm);
      float p1 = __expf(s1[r] - newm);
      s0[r] = p0; s1[r] = p1;
      rsum[r] = p0 + p1;
    }
    #pragma unroll
    for (int mstep = 1; mstep < 16; mstep <<= 1) {
      #pragma unroll
      for (int r = 0; r < 8; ++r)
        rsum[r] += __shfl_xor(rsum[r], mstep, 32);
    }
    #pragma unroll
    for (int r = 0; r < 8; ++r) lsum[r] += rsum[r];

    // ---- C-layout -> A-layout via wave-private LDS round trip ----
    #pragma unroll
    for (int r = 0; r < 8; ++r) {
      int m = r + half * 8;
      pt[m * 40 + l16]      = (__bf16)s0[r];
      pt[m * 40 + 16 + l16] = (__bf16)s1[r];
    }
    BF16x16 pa;
    pa.h[0] = *(const v8bf*)(pt + l16 * 40 + koff);
    pa.h[1] = *(const v8bf*)(pt + l16 * 40 + 8 + koff);

    // ---- O += P * V : all 4 V^T B-frags preloaded, then 4 WMMAs ----
    const __bf16* vr = &vt[buf][0] + (size_t)l16 * VPITCH + half * 16;
    BF16x16 bv0, bv1, bv2, bv3;
    bv0.h[0] = *(const v8bf*)(vr);
    bv0.h[1] = *(const v8bf*)(vr + 8);
    bv1.h[0] = *(const v8bf*)(vr + 16 * VPITCH);
    bv1.h[1] = *(const v8bf*)(vr + 16 * VPITCH + 8);
    bv2.h[0] = *(const v8bf*)(vr + 32 * VPITCH);
    bv2.h[1] = *(const v8bf*)(vr + 32 * VPITCH + 8);
    bv3.h[0] = *(const v8bf*)(vr + 48 * VPITCH);
    bv3.h[1] = *(const v8bf*)(vr + 48 * VPITCH + 8);
    o[0] = wmma_bf16(pa, bv0, o[0]);
    o[1] = wmma_bf16(pa, bv1, o[1]);
    o[2] = wmma_bf16(pa, bv2, o[2]);
    o[3] = wmma_bf16(pa, bv3, o[3]);
  }

  // ---- normalize + store row-major [b*2048+s][h*64+d] as bf16 ----
  const int b = bh >> 4, h = bh & 15;
  #pragma unroll
  for (int r = 0; r < 8; ++r) {
    float inv = 1.0f / lsum[r];
    int grow = b * SQ + qb + r + half * 8;
    __bf16* orow = attn_out + (size_t)grow * D_MODEL + h * HD + l16;
    orow[0]  = (__bf16)(o[0][r] * inv);
    orow[16] = (__bf16)(o[1][r] * inv);
    orow[32] = (__bf16)(o[2][r] * inv);
    orow[48] = (__bf16)(o[3][r] * inv);
  }
}

// ---------------------------------------------------------------------------
// Host launcher
// ---------------------------------------------------------------------------
extern "C" void kernel_launch(void* const* d_in, const int* in_sizes, int n_in,
                              void* d_out, int out_size, void* d_ws, size_t ws_size,
                              hipStream_t stream) {
  (void)in_sizes; (void)n_in; (void)out_size; (void)ws_size;
  const float* x   = (const float*)d_in[0];
  const float* kvk = (const float*)d_in[1];
  const float* kvv = (const float*)d_in[2];
  const float* WQ  = (const float*)d_in[3];
  const float* WK  = (const float*)d_in[4];
  const float* WV  = (const float*)d_in[5];
  const float* WO  = (const float*)d_in[6];

  float* out  = (float*)d_out;                                   // [2,2048,1024]
  float* outK = out  + (size_t)MROWS * D_MODEL;                  // [2,16,4096,64]
  float* outV = outK + (size_t)BATCH * NHEADS * STOT * HD;       // [2,16,4096,64]

  char* ws = (char*)d_ws;
  __bf16* xb   = (__bf16*)ws;  ws += (size_t)MROWS * D_MODEL * 2;            // 8 MB
  __bf16* WtQ  = (__bf16*)ws;  ws += (size_t)D_MODEL * D_MODEL * 2;          // 2 MB
  __bf16* WtK  = (__bf16*)ws;  ws += (size_t)D_MODEL * D_MODEL * 2;
  __bf16* WtV  = (__bf16*)ws;  ws += (size_t)D_MODEL * D_MODEL * 2;
  __bf16* WtO  = (__bf16*)ws;  ws += (size_t)D_MODEL * D_MODEL * 2;
  __bf16* Qb   = (__bf16*)ws;  ws += (size_t)BATCH * NHEADS * SQ * HD * 2;   // 8 MB
  __bf16* Kb   = (__bf16*)ws;  ws += (size_t)BATCH * NHEADS * STOT * HD * 2; // 16 MB
  __bf16* Vtb  = (__bf16*)ws;  ws += (size_t)BATCH * NHEADS * STOT * HD * 2; // 16 MB
  __bf16* attn = (__bf16*)ws;  ws += (size_t)MROWS * D_MODEL * 2;            // 8 MB

  // 1) precision conversions + weight transposes
  cvt_f32_bf16_kernel<<<(MROWS * D_MODEL / 4 + 255) / 256, 256, 0, stream>>>(x, xb, MROWS * D_MODEL / 4);
  dim3 tb(32, 8), tg(32, 32);
  transpose_cvt_kernel<<<tg, tb, 0, stream>>>(WQ, WtQ);
  transpose_cvt_kernel<<<tg, tb, 0, stream>>>(WK, WtK);
  transpose_cvt_kernel<<<tg, tb, 0, stream>>>(WV, WtV);
  transpose_cvt_kernel<<<tg, tb, 0, stream>>>(WO, WtO);

  // 2) KV-cache copy-through + bf16 staging
  cache_kernel<<<(BATCH * NHEADS * SCACHE * HD + 255) / 256, 256, 0, stream>>>(
      kvk, kvv, outK, outV, Kb, Vtb);

  // 3) Q/K/V projections (WMMA GEMMs); K/V also emit f32 into d_out regions
  dim3 gg(MROWS / 128, D_MODEL / 64);  // (32, 16)
  gemm_bf16_kernel<<<gg, 256, 0, stream>>>(xb, WtQ, Qb, 3, (float*)nullptr, 0);
  gemm_bf16_kernel<<<gg, 256, 0, stream>>>(xb, WtK, Kb, 1, outK, 1);
  gemm_bf16_kernel<<<gg, 256, 0, stream>>>(xb, WtV, Vtb, 2, outV, 1);

  // 4) attention (flash, online softmax, async-staged K/V)
  attn_kernel<<<(BATCH * NHEADS * (SQ / 16)) / 8, 256, 0, stream>>>(Qb, Kb, Vtb, attn);

  // 5) output projection -> f32
  gemm_bf16_kernel<<<gg, 256, 0, stream>>>(attn, WtO, (__bf16*)nullptr, 0, out, 2);
}
